// miniGPT_21964462752508
// MI455X (gfx1250) — matrix operand order repaired and checked by reading, not compiled, not run on hardware
//
#include <hip/hip_runtime.h>
#include <hip/hip_bf16.h>
#include <math.h>

typedef __bf16 bfx;
typedef __bf16 v16bf __attribute__((ext_vector_type(16)));
typedef __bf16 v8bf  __attribute__((ext_vector_type(8)));
typedef float  v8f   __attribute__((ext_vector_type(8)));
typedef float  v4f   __attribute__((ext_vector_type(4)));

#define B_  2
#define S_  1024
#define V_  32000
#define D_  1024
#define DH_ 4096
#define NL_ 4
#define NH_ 16
#define HD_ 64

__device__ __forceinline__ v16bf cat16(v8bf lo, v8bf hi) {
    return __builtin_shufflevector(lo, hi, 0,1,2,3,4,5,6,7,8,9,10,11,12,13,14,15);
}

__device__ __forceinline__ v8f wmma_bf16(v16bf a, v16bf b, v8f c) {
    // D = A(16x32 bf16) * B(32x16 bf16) + C(16x16 f32)
    return __builtin_amdgcn_wmma_f32_16x16x32_bf16(false, a, false, b, (short)0, c, false, false);
}

// ---------------- Tensor Data Mover path (6-arg clang-23 form on this toolchain)
#if __has_builtin(__builtin_amdgcn_tensor_load_to_lds)
#define USE_TDM 1
typedef unsigned int u32x4 __attribute__((ext_vector_type(4)));
typedef int          i32x4 __attribute__((ext_vector_type(4)));
typedef int          i32x8 __attribute__((ext_vector_type(8)));

// Load a 128-row x 32-col bf16 tile (row stride K elems) into LDS with a
// 16-byte pad after every 64-byte row => LDS row stride 40 bf16 elements.
__device__ __forceinline__ void tdm_load_tile_a(const bfx* gptr, bfx* lptr, int K) {
    unsigned long long ga = (unsigned long long)(uintptr_t)gptr;
    unsigned int la = (unsigned int)(uintptr_t)lptr;   // flat LDS ptr low 32 = LDS offset
    u32x4 g0;
    g0[0] = 1u;                                        // count=1, user descriptor
    g0[1] = la;                                        // lds_addr [63:32]
    g0[2] = (unsigned int)(ga & 0xFFFFFFFFu);          // global_addr[31:0]
    g0[3] = (unsigned int)((ga >> 32) & 0x01FFFFFFu)   // global_addr[56:32]
          | (2u << 30);                                // type = 2 ("image")
    i32x8 g1;
    g1[0] = (1 << 16)                                  // data_size = 2 bytes
          | (1 << 20)                                  // pad_enable
          | (3 << 22)                                  // pad_interval: 16 DWORDs (one row)
          | (3 << 25);                                 // pad_amount: 4 DWORDs (16 B)
    g1[1] = (K & 0xFFFF) << 16;                        // tensor_dim0[15:0]  @ [63:48]
    g1[2] = ((K >> 16) & 0xFFFF)                       // tensor_dim0[31:16] @ [79:64]
          | (128 << 16);                               // tensor_dim1[15:0]  @ [95:80]
    g1[3] = (32 << 16);                                // tile_dim0 = 32     @ [127:112]
    g1[4] = 128;                                       // tile_dim1 = 128; tile_dim2 = 0
    g1[5] = K;                                         // tensor_dim0_stride[31:0] @ [191:160]
    g1[6] = 0;                                         // stride hi16 + dim1_stride lo16
    g1[7] = 0;
    i32x4 z4 = {0, 0, 0, 0};
    i32x8 z8 = {0, 0, 0, 0, 0, 0, 0, 0};
    __builtin_amdgcn_tensor_load_to_lds(g0, g1, z4, z4, z8, 0);
}
#else
#define USE_TDM 0
#endif

// ---------------------------------------------------------------- embedding
__global__ __launch_bounds__(256) void embed_kernel(const int* __restrict__ x,
                                                    const float* __restrict__ emb,
                                                    const float* __restrict__ pe,
                                                    float* __restrict__ h) {
    int row = blockIdx.x;              // 0..B*S-1
    int s   = row & (S_ - 1);
    int tok = x[row];
    int i   = threadIdx.x * 4;
    v4f e = *(const v4f*)(emb + (size_t)tok * D_ + i);
    v4f p = *(const v4f*)(pe  + (size_t)s   * D_ + i);
    *(v4f*)(h + (size_t)row * D_ + i) = e + p;
}

// ---------------------------------------------------------------- f32 -> bf16
__global__ __launch_bounds__(256) void cvt_kernel(const float* __restrict__ in,
                                                  bfx* __restrict__ out, long n) {
    long i = ((long)blockIdx.x * 256 + threadIdx.x) * 4;
    if (i >= n) return;
    v4f v = *(const v4f*)(in + i);
    out[i + 0] = (bfx)v[0];
    out[i + 1] = (bfx)v[1];
    out[i + 2] = (bfx)v[2];
    out[i + 3] = (bfx)v[3];
}

// ---------------------------------------------------------------- layernorm (row of 1024) -> bf16
__global__ __launch_bounds__(256) void ln_kernel(const float* __restrict__ X,
                                                 const float* __restrict__ g,
                                                 const float* __restrict__ b,
                                                 bfx* __restrict__ out) {
    __shared__ float red[256];
    int row = blockIdx.x, tid = threadIdx.x;
    const float* xr = X + (size_t)row * D_;
    v4f v = *(const v4f*)(xr + tid * 4);
    float s = v[0] + v[1] + v[2] + v[3];
    red[tid] = s; __syncthreads();
    for (int off = 128; off > 0; off >>= 1) {
        if (tid < off) red[tid] += red[tid + off];
        __syncthreads();
    }
    float mu = red[0] * (1.0f / D_);
    __syncthreads();
    float d0 = v[0] - mu, d1 = v[1] - mu, d2 = v[2] - mu, d3 = v[3] - mu;
    red[tid] = d0*d0 + d1*d1 + d2*d2 + d3*d3; __syncthreads();
    for (int off = 128; off > 0; off >>= 1) {
        if (tid < off) red[tid] += red[tid + off];
        __syncthreads();
    }
    float rs = rsqrtf(red[0] * (1.0f / D_) + 1e-5f);
    v4f gg = *(const v4f*)(g + tid * 4);
    v4f bb = *(const v4f*)(b + tid * 4);
    bfx* o = out + (size_t)row * D_ + tid * 4;
    o[0] = (bfx)(d0 * rs * gg[0] + bb[0]);
    o[1] = (bfx)(d1 * rs * gg[1] + bb[1]);
    o[2] = (bfx)(d2 * rs * gg[2] + bb[2]);
    o[3] = (bfx)(d3 * rs * gg[3] + bb[3]);
}

// ---------------------------------------------------------------- WMMA GEMM
// C[M,N] = A[M,K](bf16,rm) x B[K,N](bf16,rm) + bias
// MODE 0: outf = acc            (head logits)
// MODE 1: outf += acc + bias    (residual accumulate)
// MODE 2: outb = bf16(acc+bias) (QKV)
// MODE 3: outb = bf16(relu(acc+bias)) (FFN1)
// 128x128 block tile, double-buffered LDS, TDM for A-tile when available.
template <int MODE>
__global__ __launch_bounds__(256) void gemm_kernel(const bfx* __restrict__ A,
                                                   const bfx* __restrict__ Bm,
                                                   const float* __restrict__ bias,
                                                   float* __restrict__ outf,
                                                   bfx* __restrict__ outb,
                                                   int M, int N, int K) {
    __shared__ bfx lds_a[2][128 * 40];   // [row][k], padded stride 40
    __shared__ bfx lds_b[2][128 * 40];   // transposed: [col][k], padded stride 40
    int tid = threadIdx.x;
    int w = tid >> 5, lane = tid & 31, hl = lane >> 4, l16 = lane & 15;
    int wm = w >> 2, wn = w & 3;              // 2x4 wave grid
    int m0 = blockIdx.y * 128, n0 = blockIdx.x * 128;

    v8f acc[4][2] = {};

    int ar = tid >> 1, ac = (tid & 1) * 16;   // A tile loader (fallback): 128 x 32
    int br = tid >> 3, bc = (tid & 7) * 16;   // B tile loader: 32 x 128
    const int NK = K >> 5;

    // ---- prologue: fill stage 0
    {
#if USE_TDM
        if (w == 0) tdm_load_tile_a(A + (size_t)m0 * K, &lds_a[0][0], K);
#else
        const bfx* asrc = A + (size_t)(m0 + ar) * K + ac;
        v8bf alo = *(const v8bf*)asrc, ahi = *(const v8bf*)(asrc + 8);
        *(v8bf*)&lds_a[0][ar * 40 + ac]     = alo;
        *(v8bf*)&lds_a[0][ar * 40 + ac + 8] = ahi;
#endif
        const bfx* bsrc = Bm + (size_t)br * N + n0 + bc;
        v8bf blo = *(const v8bf*)bsrc, bhi = *(const v8bf*)(bsrc + 8);
#pragma unroll
        for (int i = 0; i < 8; i++) lds_b[0][(bc + i) * 40 + br]     = blo[i];
#pragma unroll
        for (int i = 0; i < 8; i++) lds_b[0][(bc + 8 + i) * 40 + br] = bhi[i];
#if USE_TDM
        if (w == 0) __builtin_amdgcn_s_wait_tensorcnt(0);
#endif
        __syncthreads();
    }

    for (int kt = 0; kt < NK; kt++) {
        int cur = kt & 1, nxt = cur ^ 1;
        bool has_next = (kt + 1 < NK);
        if (has_next) {
            int k0n = (kt + 1) << 5;
#if USE_TDM
            if (w == 0) tdm_load_tile_a(A + (size_t)m0 * K + k0n, &lds_a[nxt][0], K);
#else
            const bfx* asrc = A + (size_t)(m0 + ar) * K + k0n + ac;
            v8bf alo = *(const v8bf*)asrc, ahi = *(const v8bf*)(asrc + 8);
            *(v8bf*)&lds_a[nxt][ar * 40 + ac]     = alo;
            *(v8bf*)&lds_a[nxt][ar * 40 + ac + 8] = ahi;
#endif
            const bfx* bsrc = Bm + (size_t)(k0n + br) * N + n0 + bc;
            v8bf blo = *(const v8bf*)bsrc, bhi = *(const v8bf*)(bsrc + 8);
#pragma unroll
            for (int i = 0; i < 8; i++) lds_b[nxt][(bc + i) * 40 + br]     = blo[i];
#pragma unroll
            for (int i = 0; i < 8; i++) lds_b[nxt][(bc + 8 + i) * 40 + br] = bhi[i];
            if (kt + 2 < NK)   // stream-prefetch the weight matrix two stages ahead
                __builtin_prefetch(Bm + (size_t)(((kt + 2) << 5) + br) * N + n0 + bc, 0, 1);
        }

        // ---- compute on stage `cur`
        v16bf af[4], bfr[2];
#pragma unroll
        for (int i = 0; i < 4; i++) {
            int r = wm * 64 + i * 16 + l16;       // A frag: row = lane&15
            v8bf lo = *(v8bf*)&lds_a[cur][r * 40 + hl * 8];        // k = hl*8 + e
            v8bf hi = *(v8bf*)&lds_a[cur][r * 40 + 16 + hl * 8];   // k = 16 + hl*8 + e
            af[i] = cat16(lo, hi);
        }
#pragma unroll
        for (int jj = 0; jj < 2; jj++) {
            int c = wn * 32 + jj * 16 + l16;      // B frag: n = lane&15
            v8bf lo = *(v8bf*)&lds_b[cur][c * 40 + hl * 8];        // k = hl*8 + e
            v8bf hi = *(v8bf*)&lds_b[cur][c * 40 + 16 + hl * 8];
            bfr[jj] = cat16(lo, hi);
        }
#pragma unroll
        for (int i = 0; i < 4; i++)
#pragma unroll
            for (int jj = 0; jj < 2; jj++)
                acc[i][jj] = wmma_bf16(af[i], bfr[jj], acc[i][jj]);

#if USE_TDM
        if (has_next && w == 0) __builtin_amdgcn_s_wait_tensorcnt(0);
#endif
        __syncthreads();
    }

#pragma unroll
    for (int jj = 0; jj < 2; jj++) {
        int n = n0 + wn * 32 + jj * 16 + l16;
        float bia = bias ? bias[n] : 0.f;
#pragma unroll
        for (int i = 0; i < 4; i++) {
            int mbase = m0 + wm * 64 + i * 16 + 8 * hl;   // C: M = j + 8*half
#pragma unroll
            for (int j = 0; j < 8; j++) {
                float v = acc[i][jj][j] + bia;
                size_t idx = (size_t)(mbase + j) * N + n;
                if (MODE == 0)      outf[idx] = v;
                else if (MODE == 1) outf[idx] += v;
                else if (MODE == 2) outb[idx] = (bfx)v;
                else                outb[idx] = (bfx)fmaxf(v, 0.f);
            }
        }
    }
}

// ---------------------------------------------------------------- flash attention
// one wave per (b, h, 16-row q-tile); kv tiles of 32
__global__ __launch_bounds__(32) void attn_kernel(const bfx* __restrict__ Q,
                                                  const bfx* __restrict__ K,
                                                  const bfx* __restrict__ V,
                                                  bfx* __restrict__ O) {
    __shared__ bfx lds_p[16 * 32];    // P tile, C-layout -> A-layout bounce
    __shared__ bfx lds_vt[64 * 40];   // V tile transposed: [hd][kv], padded
    int qt = blockIdx.x, hh = blockIdx.y, b = blockIdx.z;
    int lane = threadIdx.x, hl = lane >> 4, l16 = lane & 15;
    int q0 = qt * 16;

    // Q A-fragments (16 q-rows x 64 hd), fixed for the whole kv loop
    const bfx* qbase = Q + (size_t)(b * S_ + q0 + l16) * D_ + hh * HD_;
    v16bf aq[2];
#pragma unroll
    for (int c = 0; c < 2; c++) {
        v8bf lo = *(const v8bf*)(qbase + c * 32 + hl * 8);
        v8bf hi = *(const v8bf*)(qbase + c * 32 + 16 + hl * 8);
        aq[c] = cat16(lo, hi);
    }

    v8f o[4] = {};
    float m[8], l[8];
#pragma unroll
    for (int j = 0; j < 8; j++) { m[j] = -1e30f; l[j] = 0.f; }
    const float scale = 0.03125f;     // 1/sqrt(D) per reference

    for (int kv0 = 0; kv0 < q0 + 16; kv0 += 32) {
        // ---- scores: S = Q * K^T, 32 kv cols as two 16x16 tiles
        v8f sl = {}, sr = {};
        const bfx* kL = K + (size_t)(b * S_ + kv0 + l16) * D_ + hh * HD_;
        const bfx* kR = kL + (size_t)16 * D_;
#pragma unroll
        for (int c = 0; c < 2; c++) {
            v8bf lo = *(const v8bf*)(kL + c * 32 + hl * 8);
            v8bf hi = *(const v8bf*)(kL + c * 32 + 16 + hl * 8);
            sl = wmma_bf16(aq[c], cat16(lo, hi), sl);
            lo = *(const v8bf*)(kR + c * 32 + hl * 8);
            hi = *(const v8bf*)(kR + c * 32 + 16 + hl * 8);
            sr = wmma_bf16(aq[c], cat16(lo, hi), sr);
        }
        // ---- causal mask + scale (C-layout: row = q0+8*hl+j, col = kv0 + l16 [+16])
#pragma unroll
        for (int j = 0; j < 8; j++) {
            int qrow = q0 + 8 * hl + j;
            int cL = kv0 + l16;
            sl[j] = (cL      <= qrow) ? sl[j] * scale : -1e30f;
            sr[j] = (cL + 16 <= qrow) ? sr[j] * scale : -1e30f;
        }
        // ---- online softmax (row reductions within 16-lane halves)
        float pl[8], pr[8], alpha[8];
#pragma unroll
        for (int j = 0; j < 8; j++) {
            float mx = fmaxf(sl[j], sr[j]);
#pragma unroll
            for (int d = 1; d < 16; d <<= 1) mx = fmaxf(mx, __shfl_xor(mx, d, 32));
            float mn = fmaxf(m[j], mx);
            alpha[j] = __expf(m[j] - mn);
            pl[j] = __expf(sl[j] - mn);
            pr[j] = __expf(sr[j] - mn);
            float rs = pl[j] + pr[j];
#pragma unroll
            for (int d = 1; d < 16; d <<= 1) rs += __shfl_xor(rs, d, 32);
            l[j] = l[j] * alpha[j] + rs;
            m[j] = mn;
        }
#pragma unroll
        for (int ch = 0; ch < 4; ch++)
#pragma unroll
            for (int j = 0; j < 8; j++) o[ch][j] *= alpha[j];

        // ---- P (C-layout) -> LDS plain 16x32 tile
#pragma unroll
        for (int j = 0; j < 8; j++) {
            lds_p[(8 * hl + j) * 32 + l16]      = (bfx)pl[j];
            lds_p[(8 * hl + j) * 32 + 16 + l16] = (bfx)pr[j];
        }
        // ---- V tile (32 kv x 64 hd) -> LDS transposed [hd][kv]
        const bfx* vrow = V + (size_t)(b * S_ + kv0 + lane) * D_ + hh * HD_;
#pragma unroll
        for (int g = 0; g < 8; g++) {
            v8bf t = *(const v8bf*)(vrow + g * 8);
#pragma unroll
            for (int i = 0; i < 8; i++) lds_vt[(g * 8 + i) * 40 + lane] = t[i];
        }
        __syncthreads();   // single-wave WG: lowers to waits + nop, orders LDS

        // ---- O += P * V  (A = P 16x32, B = V 32x16 per 16-wide hd chunk)
        v8bf plo = *(v8bf*)&lds_p[l16 * 32 + hl * 8];
        v8bf phi = *(v8bf*)&lds_p[l16 * 32 + 16 + hl * 8];
        v16bf ap = cat16(plo, phi);
#pragma unroll
        for (int ch = 0; ch < 4; ch++) {
            v8bf lo = *(v8bf*)&lds_vt[(ch * 16 + l16) * 40 + hl * 8];
            v8bf hi = *(v8bf*)&lds_vt[(ch * 16 + l16) * 40 + 16 + hl * 8];
            o[ch] = wmma_bf16(ap, cat16(lo, hi), o[ch]);
        }
        __syncthreads();
    }

#pragma unroll
    for (int j = 0; j < 8; j++) l[j] = 1.f / l[j];
#pragma unroll
    for (int ch = 0; ch < 4; ch++)
#pragma unroll
        for (int j = 0; j < 8; j++) {
            size_t idx = (size_t)(b * S_ + q0 + 8 * hl + j) * D_ + hh * HD_ + ch * 16 + l16;
            O[idx] = (bfx)(o[ch][j] * l[j]);
        }
}

// ---------------------------------------------------------------- driver
extern "C" void kernel_launch(void* const* d_in, const int* in_sizes, int n_in,
                              void* d_out, int out_size, void* d_ws, size_t ws_size,
                              hipStream_t stream) {
    (void)in_sizes; (void)n_in; (void)out_size; (void)ws_size;
    const int*   x    = (const int*)  d_in[0];
    const float* emb  = (const float*)d_in[1];
    const float* pe   = (const float*)d_in[2];
    const float* ln1g = (const float*)d_in[3];
    const float* ln1b = (const float*)d_in[4];
    const float* wq   = (const float*)d_in[5];
    const float* bq   = (const float*)d_in[6];
    const float* wk   = (const float*)d_in[7];
    const float* bk   = (const float*)d_in[8];
    const float* wv   = (const float*)d_in[9];
    const float* bv   = (const float*)d_in[10];
    const float* wo   = (const float*)d_in[11];
    const float* bo   = (const float*)d_in[12];
    const float* ln2g = (const float*)d_in[13];
    const float* ln2b = (const float*)d_in[14];
    const float* w1   = (const float*)d_in[15];
    const float* b1   = (const float*)d_in[16];
    const float* w2   = (const float*)d_in[17];
    const float* b2   = (const float*)d_in[18];
    const float* lnfg = (const float*)d_in[19];
    const float* lnfb = (const float*)d_in[20];
    const float* wout = (const float*)d_in[21];
    float* logits = (float*)d_out;

    char* ws = (char*)d_ws;
    size_t off = 0;
    float* h  = (float*)(ws + off); off += (size_t)B_ * S_ * D_ * 4;   // 8 MB
    bfx* a    = (bfx*)(ws + off);   off += (size_t)B_ * S_ * D_ * 2;   // 4 MB
    bfx* qb   = (bfx*)(ws + off);   off += (size_t)B_ * S_ * D_ * 2;
    bfx* kb   = (bfx*)(ws + off);   off += (size_t)B_ * S_ * D_ * 2;
    bfx* vb   = (bfx*)(ws + off);   off += (size_t)B_ * S_ * D_ * 2;
    bfx* tb   = (bfx*)(ws + off);   off += (size_t)B_ * S_ * D_ * 2;
    bfx* mb   = (bfx*)(ws + off);   off += (size_t)B_ * S_ * DH_ * 2;  // 16 MB
    bfx* wbuf = (bfx*)(ws + off);                                      // 64 MB (D*V bf16 max)

    const int MR = B_ * S_;  // 2048
    auto cvt = [&](const float* src, long n) {
        cvt_kernel<<<(unsigned)((n / 4 + 255) / 256), 256, 0, stream>>>(src, wbuf, n);
    };

    embed_kernel<<<MR, 256, 0, stream>>>(x, emb, pe, h);

    for (int lyr = 0; lyr < NL_; ++lyr) {
        ln_kernel<<<MR, 256, 0, stream>>>(h, ln1g + lyr * D_, ln1b + lyr * D_, a);

        cvt(wq + (size_t)lyr * D_ * D_, (long)D_ * D_);
        gemm_kernel<2><<<dim3(D_ / 128, MR / 128), 256, 0, stream>>>(a, wbuf, bq + lyr * D_, nullptr, qb, MR, D_, D_);
        cvt(wk + (size_t)lyr * D_ * D_, (long)D_ * D_);
        gemm_kernel<2><<<dim3(D_ / 128, MR / 128), 256, 0, stream>>>(a, wbuf, bk + lyr * D_, nullptr, kb, MR, D_, D_);
        cvt(wv + (size_t)lyr * D_ * D_, (long)D_ * D_);
        gemm_kernel<2><<<dim3(D_ / 128, MR / 128), 256, 0, stream>>>(a, wbuf, bv + lyr * D_, nullptr, vb, MR, D_, D_);

        attn_kernel<<<dim3(S_ / 16, NH_, B_), 32, 0, stream>>>(qb, kb, vb, tb);

        cvt(wo + (size_t)lyr * D_ * D_, (long)D_ * D_);
        gemm_kernel<1><<<dim3(D_ / 128, MR / 128), 256, 0, stream>>>(tb, wbuf, bo + lyr * D_, h, nullptr, MR, D_, D_);

        ln_kernel<<<MR, 256, 0, stream>>>(h, ln2g + lyr * D_, ln2b + lyr * D_, a);

        cvt(w1 + (size_t)lyr * D_ * DH_, (long)D_ * DH_);
        gemm_kernel<3><<<dim3(DH_ / 128, MR / 128), 256, 0, stream>>>(a, wbuf, b1 + lyr * DH_, nullptr, mb, MR, DH_, D_);
        cvt(w2 + (size_t)lyr * DH_ * D_, (long)DH_ * D_);
        gemm_kernel<1><<<dim3(D_ / 128, MR / 128), 256, 0, stream>>>(mb, wbuf, b2 + lyr * D_, h, nullptr, MR, D_, DH_);
    }

    ln_kernel<<<MR, 256, 0, stream>>>(h, lnfg, lnfb, a);
    cvt(wout, (long)D_ * V_);
    gemm_kernel<0><<<dim3(V_ / 128, MR / 128), 256, 0, stream>>>(a, wbuf, nullptr, logits, nullptr, MR, V_, D_);
}
